// VALLEAR_71201967833388
// MI455X (gfx1250) — compile-verified, matrix operand order checked by reading, not run on hardware
//
#include <hip/hip_runtime.h>
#include <hip/hip_bf16.h>
#include <math.h>

typedef __bf16 bf16;
typedef __attribute__((ext_vector_type(16))) __bf16 v16bf;
typedef __attribute__((ext_vector_type(8)))  __bf16 v8bf;
typedef __attribute__((ext_vector_type(8)))  float  v8f;

constexpr int B_ = 8, TT_ = 128, TP_ = 256, TO_ = 512;
constexpr int T_ = TT_ + 1 + TP_ + 1 + TO_;   // 898
constexpr int V_ = 1024, D_ = 1024, H_ = 16, L_ = 12;
constexpr int FF_ = 4096;
constexpr int M_ = B_ * T_;                    // 7184 rows

// ---------------------------------------------------------------------------
// 16-byte global -> LDS copy via gfx1250 async LDS DMA (ASYNCcnt-tracked).
// Fallback (host pass) is a plain register round-trip.
// ---------------------------------------------------------------------------
__device__ __forceinline__ void cp16(const bf16* g, bf16* l) {
#if defined(__gfx1250__)
  asm volatile("global_load_async_to_lds_b128 %0, %1, off"
               :: "v"((unsigned)(unsigned long long)l), "v"(g)
               : "memory");
#else
  *(uint4*)l = *(const uint4*)g;
#endif
}
// Wait until at most N async loads remain outstanding (in-order completion).
template <int N>
__device__ __forceinline__ void cp16_join() {
#if defined(__gfx1250__)
  asm volatile("s_wait_asynccnt %0" :: "i"(N) : "memory");
#endif
}

// ---------------------------------------------------------------------------
// Embedding: x[b,t,d] = tab[id][d] + PE(pos,d)  (sep token gets no PE)
// ---------------------------------------------------------------------------
__global__ void embed_kernel(const int* __restrict__ text_ids,
                             const int* __restrict__ prompt_ids,
                             const int* __restrict__ output_ids,
                             const float* __restrict__ text_tab,
                             const float* __restrict__ prompt_tab,
                             const float* __restrict__ output_tab,
                             const float* __restrict__ sep,
                             float* __restrict__ x) {
  long idx = (long)blockIdx.x * blockDim.x + threadIdx.x;
  if (idx >= (long)M_ * D_) return;
  int d = (int)(idx % D_);
  long row = idx / D_;
  int t = (int)(row % T_);
  int b = (int)(row / T_);
  float val;
  int pos = 0;
  const int* ids = nullptr;
  const float* tab = nullptr;
  bool is_sep = false;
  if (t < TT_)                { pos = t;                ids = text_ids   + b * TT_; tab = text_tab; }
  else if (t == TT_)          { is_sep = true; }
  else if (t < TT_ + 1 + TP_) { pos = t - (TT_ + 1);    ids = prompt_ids + b * TP_; tab = prompt_tab; }
  else if (t == TT_ + 1 + TP_){ is_sep = true; }
  else                        { pos = t - (TT_ + 2 + TP_); ids = output_ids + b * TO_; tab = output_tab; }
  if (is_sep) {
    val = sep[d];
  } else {
    int id = ids[pos];
    float e = tab[(long)id * D_ + d];
    int dd = (d < 512) ? d : (d - 512);
    float omega = __expf(-9.210340372f * (float)dd / 512.0f);  // ln(1e4)
    float ang = (float)pos * omega;
    float pe = (d < 512) ? __sinf(ang) : __cosf(ang);
    val = e + pe;
  }
  x[idx] = val;
}

// ---------------------------------------------------------------------------
// LayerNorm(row of 1024) -> bf16
// ---------------------------------------------------------------------------
__global__ void __launch_bounds__(256)
ln_kernel(const float* __restrict__ x, const float* __restrict__ g,
          const float* __restrict__ bta, bf16* __restrict__ out) {
  __shared__ float red[256];
  int row = blockIdx.x;
  const float* xr = x + (size_t)row * D_;
  float v4[4];
  float s = 0.f;
  #pragma unroll
  for (int i = 0; i < 4; i++) { v4[i] = xr[threadIdx.x + i * 256]; s += v4[i]; }
  red[threadIdx.x] = s; __syncthreads();
  for (int off = 128; off > 0; off >>= 1) {
    if ((int)threadIdx.x < off) red[threadIdx.x] += red[threadIdx.x + off];
    __syncthreads();
  }
  float mu = red[0] * (1.0f / D_);
  __syncthreads();
  s = 0.f;
  #pragma unroll
  for (int i = 0; i < 4; i++) { float dl = v4[i] - mu; s += dl * dl; }
  red[threadIdx.x] = s; __syncthreads();
  for (int off = 128; off > 0; off >>= 1) {
    if ((int)threadIdx.x < off) red[threadIdx.x] += red[threadIdx.x + off];
    __syncthreads();
  }
  float rstd = rsqrtf(red[0] * (1.0f / D_) + 1e-5f);
  bf16* orow = out + (size_t)row * D_;
  #pragma unroll
  for (int i = 0; i < 4; i++) {
    int c = threadIdx.x + i * 256;
    orow[c] = (bf16)((v4[i] - mu) * rstd * g[c] + bta[c]);
  }
}

// ---------------------------------------------------------------------------
// fp32 KxN weight -> bf16 NxK (N-major, so B fragments read K-contiguous)
// ---------------------------------------------------------------------------
__global__ void convT_kernel(const float* __restrict__ W, bf16* __restrict__ Wt,
                             int K, int N) {
  __shared__ float tile[32][33];
  int k0 = blockIdx.y * 32, n0 = blockIdx.x * 32;
  #pragma unroll
  for (int i = 0; i < 4; i++) {
    int kk = threadIdx.y + i * 8;
    tile[kk][threadIdx.x] = W[(size_t)(k0 + kk) * N + n0 + threadIdx.x];
  }
  __syncthreads();
  #pragma unroll
  for (int i = 0; i < 4; i++) {
    int nn = threadIdx.y + i * 8;
    Wt[(size_t)(n0 + nn) * K + k0 + threadIdx.x] = (bf16)tile[threadIdx.x][nn];
  }
}

// ---------------------------------------------------------------------------
// WMMA GEMM: C = A(MxK, bf16 row-major) * B (Bt is NxK bf16, i.e. W^T)
// Block tile 128x128, K-tile 64, 8 waves (2x4), wave tile 64x32.
// Double-buffered LDS, async DMA fills overlapped with WMMA compute.
// ---------------------------------------------------------------------------
template <bool OUT_BF16, bool BIAS, bool GELU_T, bool RESID>
__global__ void __launch_bounds__(256)
gemm_bf16_kernel(const bf16* __restrict__ A, const bf16* __restrict__ Bt,
                 const float* __restrict__ bias, float* __restrict__ Xres,
                 bf16* __restrict__ OutBf, int Mrows, int N, int K) {
  __shared__ bf16 As[2][128][80];   // stride 80 bf16 = 160 B, 16B-aligned rows
  __shared__ bf16 Bs[2][128][80];
  const int tid  = threadIdx.x;
  const int lane = tid & 31;
  const int wid  = tid >> 5;
  const int wm   = wid >> 2;     // 0..1
  const int wn   = wid & 3;      // 0..3
  const int m0   = blockIdx.y * 128;
  const int n0   = blockIdx.x * 128;

  v8f acc[4][2];
  #pragma unroll
  for (int i = 0; i < 4; i++)
    #pragma unroll
    for (int j = 0; j < 2; j++)
      #pragma unroll
      for (int r = 0; r < 8; r++) acc[i][j][r] = 0.f;

  const int mr = lane & 15;
  const int kc = (lane & 16) ? 8 : 0;

  // issue one 128x64 double-tile fill (8 async b128 per wave per call)
  auto issue = [&](int kt, int buf) {
    #pragma unroll
    for (int i = 0; i < 4; i++) {
      int c = tid + i * 256;
      int r = c >> 3;
      int col = (c & 7) * 8;
      int arow = m0 + r;
      if (arow >= Mrows) arow = Mrows - 1;   // clamp: garbage row only feeds
                                             // a C-row that is never stored
      cp16(A + (size_t)arow * K + kt + col, &As[buf][r][col]);
      cp16(Bt + (size_t)(n0 + r) * K + kt + col, &Bs[buf][r][col]);
    }
  };

  const int nk = K >> 6;   // K / 64
  issue(0, 0);
  for (int it = 0; it < nk; it++) {
    const int buf = it & 1;
    if (it + 1 < nk) {
      issue((it + 1) << 6, buf ^ 1);   // prefetch next tile into other buffer
      cp16_join<8>();                  // in-order: current tile's 8 loads done
    } else {
      cp16_join<0>();
    }
    __syncthreads();                   // tile visible to all waves

    union Fr { v16bf v; v8bf h[2]; };
    #pragma unroll
    for (int s = 0; s < 2; s++) {
      Fr bfrag[2];
      #pragma unroll
      for (int j = 0; j < 2; j++) {
        const bf16* p = &Bs[buf][wn * 32 + j * 16 + mr][s * 32 + kc];
        bfrag[j].h[0] = *(const v8bf*)p;
        bfrag[j].h[1] = *(const v8bf*)(p + 16);
      }
      #pragma unroll
      for (int i = 0; i < 4; i++) {
        Fr af;
        const bf16* p = &As[buf][wm * 64 + i * 16 + mr][s * 32 + kc];
        af.h[0] = *(const v8bf*)p;
        af.h[1] = *(const v8bf*)(p + 16);
        #pragma unroll
        for (int j = 0; j < 2; j++) {
          acc[i][j] = __builtin_amdgcn_wmma_f32_16x16x32_bf16(
              false, af.v, false, bfrag[j].v, (short)0, acc[i][j], false, false);
        }
      }
    }
    __syncthreads();                   // reads done before buffer reused (it+2)
  }

  const int lm  = (lane & 16) ? 8 : 0;
  const int ln_ = lane & 15;
  #pragma unroll
  for (int i = 0; i < 4; i++) {
    #pragma unroll
    for (int j = 0; j < 2; j++) {
      int colg = n0 + wn * 32 + j * 16 + ln_;
      #pragma unroll
      for (int r = 0; r < 8; r++) {
        int rowg = m0 + wm * 64 + i * 16 + lm + r;
        if (rowg < Mrows) {
          float v = acc[i][j][r];
          if (BIAS)   v += bias[colg];
          if (GELU_T) v = 0.5f * v * (1.0f + erff(v * 0.70710678118f));
          if (RESID) {
            float* xp = Xres + (size_t)rowg * N + colg;
            *xp = *xp + v;
          }
          if (OUT_BF16) OutBf[(size_t)rowg * N + colg] = (bf16)v;
        }
      }
    }
  }
}

// ---------------------------------------------------------------------------
// WMMA flash attention. Workgroup = one (b, h, 128-query tile), 8 waves,
// each wave owns 16 query rows. 32-key tiles streamed through LDS.
// qkv row layout: [q(1024) | k(1024) | v(1024)], head h at offset h*64, DH=64.
// ---------------------------------------------------------------------------
__global__ void __launch_bounds__(256)
fattn_kernel(const bf16* __restrict__ qkv, bf16* __restrict__ obf) {
  __shared__ bf16 Ks[32][64];       // key-major K tile (B operand for Q*K^T)
  __shared__ bf16 Vt[64][48];       // d-major (transposed) V tile (B operand for P*V)
  __shared__ bf16 Ps[8][16][32];    // per-wave P tile (C-layout -> A-layout relay)

  const int tid  = threadIdx.x;
  const int lane = tid & 31;
  const int w    = tid >> 5;            // wave id 0..7
  const int bh   = blockIdx.y;
  const int h    = bh % H_;
  const int b    = bh / H_;
  const int q0   = blockIdx.x * 128;
  const int rowbase = q0 + w * 16;      // first query t of this wave
  const int mr   = lane & 15;
  const int half = (lane >> 4) & 1;
  const int kc   = half ? 8 : 0;        // ISA bf16 A/B K-chunk pattern
  const size_t baseBH = (size_t)b * T_ * 3072 + (size_t)h * 64;

  union Fr { v16bf v; v8bf h8[2]; };

  // Q fragments for the two 32-wide d-steps (A operand, loaded from global)
  Fr qf[2];
  {
    int t = rowbase + mr;
    bool ok = (t < T_);
    const bf16* qp = qkv + baseBH + (size_t)t * 3072;
    #pragma unroll
    for (int s = 0; s < 2; s++) {
      if (ok) {
        qf[s].h8[0] = *(const v8bf*)(qp + s * 32 + kc);
        qf[s].h8[1] = *(const v8bf*)(qp + s * 32 + kc + 16);
      } else {
        #pragma unroll
        for (int e = 0; e < 16; e++) qf[s].v[e] = (bf16)0.f;
      }
    }
  }

  float m[8], l[8];
  #pragma unroll
  for (int r = 0; r < 8; r++) { m[r] = -3.0e38f; l[r] = 0.f; }
  v8f o[4];
  #pragma unroll
  for (int j = 0; j < 4; j++)
    #pragma unroll
    for (int r = 0; r < 8; r++) o[j][r] = 0.f;

  const int wmax = rowbase + 15;
  const int kend = (q0 + 128 < T_) ? (q0 + 128) : T_;   // keys needed by this WG

  for (int k0 = 0; k0 < kend; k0 += 32) {
    // cooperative tile fill: K async, V via register transpose
    {
      int krow = tid >> 3;            // 0..31 key within tile
      int dcol = (tid & 7) * 8;       // 0..56
      int kt = k0 + krow;
      if (kt >= T_) kt = T_ - 1;      // clamped rows are always causally masked
      cp16(qkv + baseBH + (size_t)kt * 3072 + 1024 + dcol, &Ks[krow][dcol]);
      uint4 vv = *(const uint4*)(qkv + baseBH + (size_t)kt * 3072 + 2048 + dcol);
      const bf16* vh = (const bf16*)&vv;
      #pragma unroll
      for (int e = 0; e < 8; e++) Vt[dcol + e][krow] = vh[e];
    }
    cp16_join<0>();
    __syncthreads();

    if (k0 <= wmax) {   // wave-uniform guard
      // S = Q * K^T : 2 key-subtiles x 2 d-steps
      v8f sacc[2];
      #pragma unroll
      for (int j = 0; j < 2; j++)
        #pragma unroll
        for (int r = 0; r < 8; r++) sacc[j][r] = 0.f;
      #pragma unroll
      for (int j = 0; j < 2; j++) {
        #pragma unroll
        for (int s = 0; s < 2; s++) {
          Fr kb;
          const bf16* p = &Ks[j * 16 + mr][s * 32 + kc];
          kb.h8[0] = *(const v8bf*)p;
          kb.h8[1] = *(const v8bf*)(p + 16);
          sacc[j] = __builtin_amdgcn_wmma_f32_16x16x32_bf16(
              false, qf[s].v, false, kb.v, (short)0, sacc[j], false, false);
        }
      }
      // online softmax (per accumulator register r = one query row per half)
      const int nidx = lane & 15;
      #pragma unroll
      for (int r = 0; r < 8; r++) {
        int row = rowbase + half * 8 + r;
        float s0 = sacc[0][r] * 0.125f;   // DH^-0.5
        float s1 = sacc[1][r] * 0.125f;
        if (k0 + nidx > row)      s0 = -3.0e38f;
        if (k0 + 16 + nidx > row) s1 = -3.0e38f;
        float mx = fmaxf(s0, s1);
        #pragma unroll
        for (int d = 1; d < 16; d <<= 1) mx = fmaxf(mx, __shfl_xor(mx, d, 32));
        float mn = fmaxf(m[r], mx);
        float corr = __expf(m[r] - mn);
        float p0 = __expf(s0 - mn);
        float p1 = __expf(s1 - mn);
        float ps = p0 + p1;
        #pragma unroll
        for (int d = 1; d < 16; d <<= 1) ps += __shfl_xor(ps, d, 32);
        l[r] = l[r] * corr + ps;
        m[r] = mn;
        #pragma unroll
        for (int j = 0; j < 4; j++) o[j][r] *= corr;
        Ps[w][half * 8 + r][nidx]      = (bf16)p0;
        Ps[w][half * 8 + r][16 + nidx] = (bf16)p1;
      }
      // O += P * V  (P relayed via per-wave LDS: C-layout -> A-layout)
      Fr pa;
      const bf16* pp = &Ps[w][mr][0];
      pa.h8[0] = *(const v8bf*)(pp + kc);
      pa.h8[1] = *(const v8bf*)(pp + kc + 16);
      #pragma unroll
      for (int j = 0; j < 4; j++) {
        Fr vb;
        const bf16* vp2 = &Vt[j * 16 + mr][kc];
        vb.h8[0] = *(const v8bf*)vp2;
        vb.h8[1] = *(const v8bf*)(vp2 + 16);
        o[j] = __builtin_amdgcn_wmma_f32_16x16x32_bf16(
            false, pa.v, false, vb.v, (short)0, o[j], false, false);
      }
    }
    __syncthreads();
  }

  // write O / l
  const int ln_ = lane & 15;
  #pragma unroll
  for (int r = 0; r < 8; r++) {
    int t = rowbase + half * 8 + r;
    if (t < T_) {
      float inv = 1.0f / l[r];
      bf16* op = obf + ((size_t)(b * T_ + t)) * 1024 + h * 64;
      #pragma unroll
      for (int j = 0; j < 4; j++)
        op[j * 16 + ln_] = (bf16)(o[j][r] * inv);
    }
  }
}

// ---------------------------------------------------------------------------
// Final logits: out[b,v] = dot(x[b,T-1,:], fc_w[:,v]) + fc_b[v]
// ---------------------------------------------------------------------------
__global__ void __launch_bounds__(256)
logits_kernel(const float* __restrict__ x, const float* __restrict__ fcw,
              const float* __restrict__ fcb, float* __restrict__ out) {
  int idx = blockIdx.x * 256 + threadIdx.x;
  int v = idx % V_;
  int b = idx / V_;
  const float* xr = x + ((size_t)(b * T_ + T_ - 1)) * D_;
  float s = 0.f;
  for (int d = 0; d < D_; d++) s += xr[d] * fcw[(size_t)d * V_ + v];
  out[idx] = s + fcb[v];
}

// ---------------------------------------------------------------------------
extern "C" void kernel_launch(void* const* d_in, const int* in_sizes, int n_in,
                              void* d_out, int out_size, void* d_ws, size_t ws_size,
                              hipStream_t stream) {
  (void)in_sizes; (void)n_in; (void)out_size; (void)ws_size;
  const int*   text_ids   = (const int*)  d_in[0];
  const int*   prompt_ids = (const int*)  d_in[1];
  const int*   output_ids = (const int*)  d_in[2];
  const float* text_tab   = (const float*)d_in[3];
  const float* prompt_tab = (const float*)d_in[4];
  const float* output_tab = (const float*)d_in[5];
  const float* sep        = (const float*)d_in[6];
  const float* ln1_g      = (const float*)d_in[7];
  const float* ln1_b      = (const float*)d_in[8];
  const float* qkv_w      = (const float*)d_in[9];
  const float* out_w      = (const float*)d_in[10];
  const float* out_b      = (const float*)d_in[11];
  const float* ln2_g      = (const float*)d_in[12];
  const float* ln2_b      = (const float*)d_in[13];
  const float* ff1_w      = (const float*)d_in[14];
  const float* ff1_b      = (const float*)d_in[15];
  const float* ff2_w      = (const float*)d_in[16];
  const float* ff2_b      = (const float*)d_in[17];
  const float* fc_w       = (const float*)d_in[18];
  const float* fc_b       = (const float*)d_in[19];

  char* ws = (char*)d_ws;
  size_t off = 0;
  auto alloc = [&](size_t bytes) -> char* {
    char* p = ws + off;
    off += (bytes + 255) & ~(size_t)255;
    return p;
  };
  float* xf    = (float*)alloc((size_t)M_ * D_ * 4);
  bf16*  hbf   = (bf16*) alloc((size_t)M_ * D_ * 2);
  bf16*  qkvbf = (bf16*) alloc((size_t)M_ * 3 * D_ * 2);
  bf16*  obf   = (bf16*) alloc((size_t)M_ * D_ * 2);
  bf16*  ffbf  = (bf16*) alloc((size_t)M_ * FF_ * 2);
  bf16*  wqkvT = (bf16*) alloc((size_t)3 * D_ * D_ * 2);
  bf16*  woutT = (bf16*) alloc((size_t)D_ * D_ * 2);
  bf16*  wff1T = (bf16*) alloc((size_t)FF_ * D_ * 2);
  bf16*  wff2T = (bf16*) alloc((size_t)D_ * FF_ * 2);

  // 1) embeddings + PE
  {
    long total = (long)M_ * D_;
    int blocks = (int)((total + 255) / 256);
    embed_kernel<<<blocks, 256, 0, stream>>>(text_ids, prompt_ids, output_ids,
                                             text_tab, prompt_tab, output_tab,
                                             sep, xf);
  }

  const int mtiles = (M_ + 127) / 128;   // 57
  dim3 tb(32, 8);
  for (int l = 0; l < L_; l++) {
    // per-layer weight convert+transpose (fp32 -> bf16, N-major)
    convT_kernel<<<dim3(3 * D_ / 32, D_ / 32), tb, 0, stream>>>(
        qkv_w + (size_t)l * D_ * 3 * D_, wqkvT, D_, 3 * D_);
    convT_kernel<<<dim3(D_ / 32, D_ / 32), tb, 0, stream>>>(
        out_w + (size_t)l * D_ * D_, woutT, D_, D_);
    convT_kernel<<<dim3(FF_ / 32, D_ / 32), tb, 0, stream>>>(
        ff1_w + (size_t)l * D_ * FF_, wff1T, D_, FF_);
    convT_kernel<<<dim3(D_ / 32, FF_ / 32), tb, 0, stream>>>(
        ff2_w + (size_t)l * FF_ * D_, wff2T, FF_, D_);

    // attention block
    ln_kernel<<<M_, 256, 0, stream>>>(xf, ln1_g + (size_t)l * D_, ln1_b + (size_t)l * D_, hbf);
    gemm_bf16_kernel<true, false, false, false>
        <<<dim3(3 * D_ / 128, mtiles), 256, 0, stream>>>(
            hbf, wqkvT, nullptr, nullptr, qkvbf, M_, 3 * D_, D_);
    fattn_kernel<<<dim3((T_ + 127) / 128, B_ * H_), 256, 0, stream>>>(qkvbf, obf);
    gemm_bf16_kernel<false, true, false, true>
        <<<dim3(D_ / 128, mtiles), 256, 0, stream>>>(
            obf, woutT, out_b + (size_t)l * D_, xf, nullptr, M_, D_, D_);

    // FFN block
    ln_kernel<<<M_, 256, 0, stream>>>(xf, ln2_g + (size_t)l * D_, ln2_b + (size_t)l * D_, hbf);
    gemm_bf16_kernel<true, true, true, false>
        <<<dim3(FF_ / 128, mtiles), 256, 0, stream>>>(
            hbf, wff1T, ff1_b + (size_t)l * FF_, nullptr, ffbf, M_, FF_, D_);
    gemm_bf16_kernel<false, true, false, true>
        <<<dim3(D_ / 128, mtiles), 256, 0, stream>>>(
            ffbf, wff2T, ff2_b + (size_t)l * D_, xf, nullptr, M_, D_, FF_);
  }

  // 3) last-position logits
  logits_kernel<<<(B_ * V_) / 256, 256, 0, stream>>>(xf, fc_w, fc_b, (float*)d_out);
}
